// SoftKNNLayer_3058016714928
// MI455X (gfx1250) — matrix operand order repaired and checked by reading, not compiled, not run on hardware
//
#include <hip/hip_runtime.h>

typedef __attribute__((ext_vector_type(16))) _Float16 v16h;
typedef __attribute__((ext_vector_type(8)))  _Float16 v8h;
typedef __attribute__((ext_vector_type(4)))  _Float16 v4h;
typedef __attribute__((ext_vector_type(8)))  float    v8f;

#define B_Q   2048
#define N_S   131072
#define DIM   128
#define NS    32                 // N-chunks per row
#define CHUNK (N_S / NS)         // 4096 columns per wave
#define TILES (CHUNK / 16)       // 256 tiles per wave
#define CAP   64                 // per-row candidate buffer capacity
#define WPB   2                  // waves per block (topk kernel)
#define TSTRIDE (16 * DIM)       // halves per 16-column tile

// ---------------------------------------------------------------------------
// Kernel 1: convert support to fp16, compute exact f32 row norms ||s||^2
// ---------------------------------------------------------------------------
__global__ void sk_prep(const float* __restrict__ s,
                        _Float16* __restrict__ sf,
                        float* __restrict__ ssq)
{
    const int lane = threadIdx.x & 31;
    const int row  = (int)((blockIdx.x * blockDim.x + threadIdx.x) >> 5);
    const float4 v = *(const float4*)(s + (size_t)row * DIM + lane * 4);
    float sum = v.x * v.x + v.y * v.y + v.z * v.z + v.w * v.w;
#pragma unroll
    for (int o = 16; o; o >>= 1) sum += __shfl_xor(sum, o, 32);
    v4h h;
    h[0] = (_Float16)v.x; h[1] = (_Float16)v.y;
    h[2] = (_Float16)v.z; h[3] = (_Float16)v.w;
    *(v4h*)(sf + (size_t)row * DIM + lane * 4) = h;
    if (lane == 0) ssq[row] = sum;
}

// ---------------------------------------------------------------------------
// wave-local selection: keep 32 smallest (d, idx) of c (<= CAP) candidates.
// ---------------------------------------------------------------------------
__device__ __forceinline__ void compact_row(float* cd, int* ci, int c, int lane,
                                            float* tau_slot, unsigned* cnt_slot)
{
    float d0 = 3.4e38f, d1 = 3.4e38f;
    int   i0 = 0x7fffffff, i1 = 0x7fffffff;
    const bool v0 = lane < c;
    const bool v1 = (lane + 32) < c;
    if (v0) { d0 = cd[lane];      i0 = ci[lane];      }
    if (v1) { d1 = cd[lane + 32]; i1 = ci[lane + 32]; }
    int r0 = 0, r1 = 0;
    for (int j = 0; j < c; ++j) {
        const float dj = cd[j]; const int ij = ci[j];
        r0 += (dj < d0) || ((dj == d0) && (ij < i0));
        r1 += (dj < d1) || ((dj == d1) && (ij < i1));
    }
    float keep = -3.4e38f;
    if (v0 && r0 < 32) { cd[r0] = d0; ci[r0] = i0; keep = d0; }
    if (v1 && r1 < 32) { cd[r1] = d1; ci[r1] = i1; keep = fmaxf(keep, d1); }
#pragma unroll
    for (int o = 16; o; o >>= 1) keep = fmaxf(keep, __shfl_xor(keep, o, 32));
    if (lane == 0) { *tau_slot = keep; *cnt_slot = 32u; }
}

// ---------------------------------------------------------------------------
// Kernel 2: fused  v = ||s||^2 - 2 x.s  (WMMA f16 -> f32) + streaming top-32
// __launch_bounds__(64,1): full VGPR budget -> no scratch spills of the
// double-buffered fragments. Running pointers avoid per-tile 64-bit address
// recomputation. Filter is one wave-uniform ballot; fast path has no DS ops.
// ---------------------------------------------------------------------------
__global__ void __launch_bounds__(WPB * 32, 1)
sk_topk(const float* __restrict__ x,
        const _Float16* __restrict__ sf,
        const float* __restrict__ ssq,
        float* __restrict__ outd,
        int* __restrict__ outi)
{
    __shared__ float    s_cd[WPB][32][CAP];
    __shared__ int      s_ci[WPB][32][CAP];
    __shared__ unsigned s_cnt[WPB][32];
    __shared__ float    s_tau[WPB][32];

    const int lane    = threadIdx.x & 31;
    const int wv      = threadIdx.x >> 5;
    const int gw      = blockIdx.x * WPB + wv;
    const int mtile   = gw / NS;
    const int nchunk  = gw % NS;
    const int b0      = mtile * 32;
    const int nbase   = nchunk * CHUNK;
    const int lane16  = lane & 15;
    const int half8   = (lane >> 4) * 8;   // A-fragment K sub-offset
    const int khalf16 = (lane >> 4) * 8;   // B-fragment K sub-offset

    s_cnt[wv][lane] = 0u;
    s_tau[wv][lane] = 3.4e38f;

    // Load A fragments (x rows, f32 -> f16), resident for whole kernel.
    v16h A[2][4];
#pragma unroll
    for (int s = 0; s < 2; ++s) {
        const float* xr = x + (size_t)(b0 + s * 16 + lane16) * DIM;
#pragma unroll
        for (int kc = 0; kc < 4; ++kc) {
            v16h a;
#pragma unroll
            for (int j = 0; j < 8; ++j) {
                a[j]     = (_Float16)xr[kc * 32 + half8 + j];
                a[j + 8] = (_Float16)xr[kc * 32 + 16 + half8 + j];
            }
            A[s][kc] = a;
        }
    }

    float tau_r[2][8];
#pragma unroll
    for (int s = 0; s < 2; ++s)
#pragma unroll
        for (int r = 0; r < 8; ++r) tau_r[s][r] = 3.4e38f;

    auto load_tile = [&](const _Float16* bp, v8h* l, v8h* h) {
#pragma unroll
        for (int kc = 0; kc < 4; ++kc) {
            l[kc] = *(const v8h*)(bp + kc * 32);
            h[kc] = *(const v8h*)(bp + kc * 32 + 16);
        }
    };

    auto push = [&](int m, float v, int n) {
        const unsigned p = atomicAdd(&s_cnt[wv][m], 1u);
        if (p < CAP) { s_cd[wv][m][p] = v; s_ci[wv][m][p] = n; }
    };

    auto process = [&](int t, const v8h* bl, const v8h* bh, float sq) {
        v8f acc0 = {}, acc1 = {};
#pragma unroll
        for (int kc = 0; kc < 4; ++kc) {
            const v16h bm = __builtin_shufflevector(bl[kc], bh[kc],
                0, 1, 2, 3, 4, 5, 6, 7, 8, 9, 10, 11, 12, 13, 14, 15);
            acc0 = __builtin_amdgcn_wmma_f32_16x16x32_f16(
                false, A[0][kc], false, bm, (short)0, acc0, false, false);
            acc1 = __builtin_amdgcn_wmma_f32_16x16x32_f16(
                false, A[1][kc], false, bm, (short)0, acc1, false, false);
        }
        const int ncol = nbase + t * 16 + lane16;
        float va[8], vb[8];
        bool anyp = false;
#pragma unroll
        for (int r = 0; r < 8; ++r) {
            va[r] = fmaf(-2.0f, acc0[r], sq);
            vb[r] = fmaf(-2.0f, acc1[r], sq);
            anyp = anyp | (va[r] < tau_r[0][r]) | (vb[r] < tau_r[1][r]);
        }
        if (__ballot(anyp) != 0ull) {         // rare path
#pragma unroll
            for (int r = 0; r < 8; ++r) {
                if (va[r] < tau_r[0][r]) push(half8 + r,      va[r], ncol);
                if (vb[r] < tau_r[1][r]) push(16 + half8 + r, vb[r], ncol);
            }
            const unsigned c = s_cnt[wv][lane];
            if (__ballot(c > (unsigned)(CAP - 16)) != 0ull) {
#pragma unroll 1
                for (int m = 0; m < 32; ++m) {
                    unsigned cm = s_cnt[wv][m];
                    if (cm > 32u) {
                        if (cm > (unsigned)CAP) cm = CAP;
                        compact_row(&s_cd[wv][m][0], &s_ci[wv][m][0], (int)cm,
                                    lane, &s_tau[wv][m], &s_cnt[wv][m]);
                    }
                }
#pragma unroll
                for (int s = 0; s < 2; ++s)
#pragma unroll
                    for (int r = 0; r < 8; ++r)
                        tau_r[s][r] = s_tau[wv][s * 16 + half8 + r];
            }
        }
    };

    // software-pipelined main loop (double-buffered B fragments,
    // running pointers, last 2-tile iteration peeled)
    v8h bAl[4], bAh[4], bBl[4], bBh[4];
    float sqA, sqB;
    const _Float16* p   = sf + (size_t)(nbase + lane16) * DIM + khalf16;
    const float*    sqp = ssq + nbase + lane16;

    load_tile(p, bAl, bAh);
    sqA = sqp[0];

    for (int t = 0; t < TILES - 2; t += 2) {
        load_tile(p + TSTRIDE, bBl, bBh);
        sqB = sqp[16];

        __builtin_prefetch((const char*)(p + 2 * TSTRIDE), 0, 1);
        __builtin_prefetch((const char*)(p + 2 * TSTRIDE) + 128, 0, 1);

        process(t, bAl, bAh, sqA);

        load_tile(p + 2 * TSTRIDE, bAl, bAh);
        sqA = sqp[32];

        process(t + 1, bBl, bBh, sqB);

        p   += 2 * TSTRIDE;
        sqp += 32;
    }
    // epilogue: tiles TILES-2, TILES-1
    load_tile(p + TSTRIDE, bBl, bBh);
    sqB = sqp[16];
    process(TILES - 2, bAl, bAh, sqA);
    process(TILES - 1, bBl, bBh, sqB);

    // final compaction + emit exactly 32 candidates per (row, chunk)
#pragma unroll 1
    for (int m = 0; m < 32; ++m) {
        unsigned cm = s_cnt[wv][m];
        if (cm > 32u) {
            if (cm > (unsigned)CAP) cm = CAP;
            compact_row(&s_cd[wv][m][0], &s_ci[wv][m][0], (int)cm,
                        lane, &s_tau[wv][m], &s_cnt[wv][m]);
        }
    }
#pragma unroll 1
    for (int m = 0; m < 32; ++m) {
        const size_t o = (((size_t)(b0 + m)) * NS + nchunk) * 32 + lane;
        outd[o] = s_cd[wv][m][lane];
        outi[o] = s_ci[wv][m][lane];
    }
}

// ---------------------------------------------------------------------------
// Kernel 3: per-row merge 32x32 -> 128 -> 32, then softmax -> class probs
// ---------------------------------------------------------------------------
__global__ void sk_finalize(const float* __restrict__ x,
                            const int* __restrict__ labels,
                            const float* __restrict__ temp,
                            const float* __restrict__ cd,
                            const int* __restrict__ ci,
                            float* __restrict__ out)
{
    __shared__ float ld[NS * 32];
    __shared__ int   li[NS * 32];
    __shared__ float md[128];
    __shared__ int   mi[128];

    const int b    = blockIdx.x;
    const int tid  = threadIdx.x;
    const int lane = tid & 31;
    const int wv   = tid >> 5;
    const size_t base = (size_t)b * (NS * 32);

    for (int j = tid; j < NS * 32; j += 128) { ld[j] = cd[base + j]; li[j] = ci[base + j]; }
    __syncthreads();

    {   // stage A: each wave selects top-32 of its 256-candidate segment
        const int seg = wv * 256;
        float dv[8]; int iv[8]; int rk[8];
#pragma unroll
        for (int j = 0; j < 8; ++j) {
            const int e = seg + lane + 32 * j;
            dv[j] = ld[e]; iv[j] = li[e]; rk[j] = 0;
        }
        for (int i = 0; i < 256; ++i) {
            const float di = ld[seg + i]; const int ii = li[seg + i];
#pragma unroll
            for (int j = 0; j < 8; ++j)
                rk[j] += (di < dv[j]) || ((di == dv[j]) && (ii < iv[j]));
        }
#pragma unroll
        for (int j = 0; j < 8; ++j)
            if (rk[j] < 32) { md[wv * 32 + rk[j]] = dv[j]; mi[wv * 32 + rk[j]] = iv[j]; }
    }
    __syncthreads();

    if (wv == 0) {
        float dv[4]; int iv[4]; int rk[4];
#pragma unroll
        for (int j = 0; j < 4; ++j) {
            const int e = lane + 32 * j;
            dv[j] = md[e]; iv[j] = mi[e]; rk[j] = 0;
        }
        for (int i = 0; i < 128; ++i) {
            const float di = md[i]; const int ii = mi[i];
#pragma unroll
            for (int j = 0; j < 4; ++j)
                rk[j] += (di < dv[j]) || ((di == dv[j]) && (ii < iv[j]));
        }

        const float4 xv = *(const float4*)(x + (size_t)b * DIM + lane * 4);
        float xs = xv.x * xv.x + xv.y * xv.y + xv.z * xv.z + xv.w * xv.w;
#pragma unroll
        for (int o = 16; o; o >>= 1) xs += __shfl_xor(xs, o, 32);

        const float T = *temp;

        float dist[4]; bool win[4];
        float dmin = 3.4e38f;
#pragma unroll
        for (int j = 0; j < 4; ++j) {
            win[j]  = rk[j] < 32;
            dist[j] = sqrtf(fmaxf(xs + dv[j], 0.0f));
            if (win[j]) dmin = fminf(dmin, dist[j]);
        }
#pragma unroll
        for (int o = 16; o; o >>= 1) dmin = fminf(dmin, __shfl_xor(dmin, o, 32));

        float a0 = 0.f, a1 = 0.f, a2 = 0.f, a3 = 0.f, W = 0.f;
#pragma unroll
        for (int j = 0; j < 4; ++j) {
            if (win[j]) {
                const float wgt = expf((dmin - dist[j]) / T);
                W += wgt;
                const int lab = labels[iv[j]];
                a0 += (lab == 0) ? wgt : 0.f;
                a1 += (lab == 1) ? wgt : 0.f;
                a2 += (lab == 2) ? wgt : 0.f;
                a3 += (lab == 3) ? wgt : 0.f;
            }
        }
#pragma unroll
        for (int o = 16; o; o >>= 1) {
            W  += __shfl_xor(W,  o, 32);
            a0 += __shfl_xor(a0, o, 32);
            a1 += __shfl_xor(a1, o, 32);
            a2 += __shfl_xor(a2, o, 32);
            a3 += __shfl_xor(a3, o, 32);
        }
        if (lane == 0) {
            const float inv = 1.0f / W;
            float* o4 = out + (size_t)b * 4;
            o4[0] = a0 * inv; o4[1] = a1 * inv; o4[2] = a2 * inv; o4[3] = a3 * inv;
        }
    }
}

// ---------------------------------------------------------------------------
extern "C" void kernel_launch(void* const* d_in, const int* in_sizes, int n_in,
                              void* d_out, int out_size, void* d_ws, size_t ws_size,
                              hipStream_t stream)
{
    const float* x      = (const float*)d_in[0];
    const float* sup    = (const float*)d_in[1];
    const int*   labels = (const int*)d_in[2];
    const float* temp   = (const float*)d_in[3];
    float*       out    = (float*)d_out;

    // workspace layout:
    //   sf16 : N*D f16          = 32 MB
    //   ssq  : N   f32          = 0.5 MB
    //   cd   : B*NS*32 f32      = 8 MB
    //   ci   : B*NS*32 i32      = 8 MB
    char* w = (char*)d_ws;
    _Float16* sf  = (_Float16*)w;
    size_t off = (size_t)N_S * DIM * sizeof(_Float16);
    float* ssq = (float*)(w + off);            off += (size_t)N_S * sizeof(float);
    float* cd  = (float*)(w + off);            off += (size_t)B_Q * NS * 32 * sizeof(float);
    int*   ci  = (int*)(w + off);

    sk_prep<<<N_S / 8, 256, 0, stream>>>(sup, sf, ssq);
    sk_topk<<<(B_Q / 32) * NS / WPB, WPB * 32, 0, stream>>>(x, sf, ssq, cd, ci);
    sk_finalize<<<B_Q, 128, 0, stream>>>(x, labels, temp, cd, ci, out);
}
